// LshAttention_9483287790113
// MI455X (gfx1250) — matrix operand order repaired and checked by reading, not compiled, not run on hardware
//
#include <hip/hip_runtime.h>
#include <hip/hip_bf16.h>

typedef __attribute__((ext_vector_type(16))) _Float16 v16h;
typedef __attribute__((ext_vector_type(8)))  float    v8f;
typedef __attribute__((ext_vector_type(4)))  int      v4i;

static constexpr int Bb  = 8;
static constexpr int S   = 8192;
static constexpr int D   = 64;
static constexpr int NH  = 8;
static constexpr int BS  = 16;
static constexpr int NB  = S / BS;      // 512 buckets per hash
static constexpr int NB2 = NB / 2;      // 256 rotation cols
static constexpr int HS  = NH * S;      // 65536 flat tokens per batch
static constexpr int NC  = NH * NB;     // 4096 chunks per batch
static constexpr int BINS = 16384;      // sort key range (bucket + pos < 12288)
static constexpr int LSTRIDE = 68;      // LDS row stride (floats): 16B aligned, banks spread
#define SELF_VAL (-50000.0f)

// ---- gfx1250 async global->LDS path (guarded; fallback keeps compile safe) ----
#if defined(__has_builtin)
#  if __has_builtin(__builtin_amdgcn_global_load_async_to_lds_b128)
#    define HAVE_ASYNC_LDS 1
#  endif
#endif
#ifndef HAVE_ASYNC_LDS
#  define HAVE_ASYNC_LDS 0
#endif

#if HAVE_ASYNC_LDS
// builtin signature (from hipcc diagnostic): (v4i as1*, v4i as3*, Ii offset, Ii cpol)
#  define AS1V(p) ((__attribute__((address_space(1))) v4i*)(p))
#  define AS3V(p) ((__attribute__((address_space(3))) v4i*)(p))
#  if __has_builtin(__builtin_amdgcn_s_wait_asynccnt)
#    define ASYNC_WAIT() __builtin_amdgcn_s_wait_asynccnt(0)
#  else
#    define ASYNC_WAIT() asm volatile("s_wait_asynccnt 0" ::: "memory")
#  endif
#endif

// A-matrix (16x32 f16) K index for element e, lane-half kh
__device__ __forceinline__ int kmapA(int e, int kh) { return (e & 7) + ((e >> 3) << 4) + (kh << 3); }
// B-matrix (32x16 f16) K index
__device__ __forceinline__ int kmapB(int e, int kh) { return (e & 7) + ((e >> 3) << 4) + (kh << 3); }

// ---------------------------------------------------------------------------
// 1) Hash: rx = qk @ rotations (per b,hash), bucket = argmax([rx, -rx])
//    One wave per (b, hash, 16-row tile). 32 WMMAs per wave.
// ---------------------------------------------------------------------------
__global__ void __launch_bounds__(32)
hash_kernel(const float* __restrict__ qk, const float* __restrict__ rot,
            int* __restrict__ keys, int* __restrict__ out_buckets)
{
    int blk  = blockIdx.x;
    int tile = blk & (S / 16 - 1);      // 512
    int g    = (blk >> 9) & (NH - 1);   // 8
    int bi   = blk >> 12;
    int lane = threadIdx.x;
    int nidx = lane & 15, kh = lane >> 4;

    // A: qk rows tile*16 + 0..15, all lanes carry row nidx (halves split K)
    const float* qrow = qk + ((size_t)bi * S + (size_t)(tile * 16 + nidx)) * D;
    v16h a0, a1;
    for (int e = 0; e < 16; ++e) {
        a0[e] = (_Float16)qrow[kmapA(e, kh)];
        a1[e] = (_Float16)qrow[32 + kmapA(e, kh)];
    }

    float bmax[8], bmin[8]; int bmaxi[8], bmini[8];
    for (int r = 0; r < 8; ++r) { bmax[r] = -3.4e38f; bmin[r] = 3.4e38f; bmaxi[r] = 0; bmini[r] = 0; }

    // rotations: (b, f, h, i) -> rot[((bi*D + f)*NH + g)*NB2 + i]
    const float* rbase = rot + (((size_t)bi * D) * NH + (size_t)g) * NB2;
    const size_t fstride = (size_t)NH * NB2;

    for (int t = 0; t < NB2 / 16; ++t) {           // 16 column tiles
        int n = t * 16 + nidx;
        v16h b0, b1;
        for (int e = 0; e < 16; ++e) {
            int f0 = kmapB(e, kh);
            b0[e] = (_Float16)rbase[(size_t)f0 * fstride + n];
            b1[e] = (_Float16)rbase[(size_t)(f0 + 32) * fstride + n];
        }
        v8f c = {};
        c = __builtin_amdgcn_wmma_f32_16x16x32_f16(false, a0, false, b0, (short)0, c, false, false);
        c = __builtin_amdgcn_wmma_f32_16x16x32_f16(false, a1, false, b1, (short)0, c, false, false);
        for (int r = 0; r < 8; ++r) {
            float val = c[r];
            if (val > bmax[r] || (val == bmax[r] && n < bmaxi[r])) { bmax[r] = val; bmaxi[r] = n; }
            if (val < bmin[r] || (val == bmin[r] && n < bmini[r])) { bmin[r] = val; bmini[r] = n; }
        }
    }

    // reduce over the 16 lanes of each half (rows differ between halves)
    for (int r = 0; r < 8; ++r) {
        float mv = bmax[r]; int mi = bmaxi[r];
        float nv = bmin[r]; int ni = bmini[r];
        for (int off = 1; off < 16; off <<= 1) {
            float omv = __shfl_xor(mv, off, 32); int omi = __shfl_xor(mi, off, 32);
            if (omv > mv || (omv == mv && omi < mi)) { mv = omv; mi = omi; }
            float onv = __shfl_xor(nv, off, 32); int oni = __shfl_xor(ni, off, 32);
            if (onv < nv || (onv == nv && oni < ni)) { nv = onv; ni = oni; }
        }
        if (nidx == 0) {
            int idx = (mv >= -nv) ? mi : (NB2 + ni);   // argmax over [rx, -rx]
            int m = r + (kh << 3);
            int p = tile * 16 + m;
            int bucket = g * NB + idx;
            int tflat = g * S + p;
            out_buckets[(size_t)bi * HS + tflat] = bucket;
            keys[(size_t)bi * HS + tflat] = bucket + p;   // sort key (constant +s dropped)
        }
    }
}

// ---------------------------------------------------------------------------
// 2) Counting sort per batch: zero -> histogram -> scan -> scatter
// ---------------------------------------------------------------------------
__global__ void zero_hist(int* __restrict__ hist)
{
    int i = blockIdx.x * blockDim.x + threadIdx.x;
    if (i < Bb * BINS) hist[i] = 0;
}

__global__ void hist_kernel(const int* __restrict__ keys, int* __restrict__ hist)
{
    int i = blockIdx.x * blockDim.x + threadIdx.x;
    if (i < Bb * HS) {
        int bi = i >> 16;                          // HS == 65536
        atomicAdd(&hist[bi * BINS + keys[i]], 1);
    }
}

__global__ void __launch_bounds__(256) scan_kernel(int* __restrict__ hist)
{
    __shared__ int tot[256];
    int bi = blockIdx.x, tid = threadIdx.x;
    int base = bi * BINS + tid * 64;
    int s = 0;
    for (int i = 0; i < 64; ++i) s += hist[base + i];
    tot[tid] = s;
    __syncthreads();
    if (tid == 0) {
        int run = 0;
        for (int i = 0; i < 256; ++i) { int t = tot[i]; tot[i] = run; run += t; }
    }
    __syncthreads();
    int run = tot[tid];
    for (int i = 0; i < 64; ++i) { int c = hist[base + i]; hist[base + i] = run; run += c; }
}

__global__ void scatter_kernel(const int* __restrict__ keys, int* __restrict__ hist,
                               int* __restrict__ sticker, int* __restrict__ inv)
{
    int i = blockIdx.x * blockDim.x + threadIdx.x;
    if (i < Bb * HS) {
        int bi = i >> 16;
        int tl = i & (HS - 1);
        int slot = atomicAdd(&hist[bi * BINS + keys[i]], 1);
        sticker[(size_t)bi * HS + slot] = tl;
        inv[(size_t)bi * HS + tl] = slot;
    }
}

// ---------------------------------------------------------------------------
// 3) Chunk attention: one wave per (b, chunk). 8 WMMAs per wave.
//    Row gather goes through GLOBAL_LOAD_ASYNC_TO_LDS_B128 (ASYNCcnt path).
// ---------------------------------------------------------------------------
__global__ void __launch_bounds__(32)
attn_kernel(const float* __restrict__ qk, const float* __restrict__ vin,
            const int* __restrict__ sticker,
            float* __restrict__ so, float* __restrict__ slog)
{
    __shared__ float    sQK[32][LSTRIDE];
    __shared__ float    sV [32][LSTRIDE];
    __shared__ float    invn[32];
    __shared__ int      pos[32];
    __shared__ _Float16 sP[16][32];

    int blk  = blockIdx.x;
    int c    = blk & (NC - 1);          // 4096
    int bi   = blk >> 12;
    int lane = threadIdx.x;
    int nidx = lane & 15, kh = lane >> 4;

    // rows 0..15: current chunk; rows 16..31: previous chunk (look-one-back, wrap)
    int cc   = (lane < 16) ? c : ((c + NC - 1) & (NC - 1));
    int tl   = sticker[(size_t)bi * HS + cc * 16 + nidx];
    int p    = tl & (S - 1);
    pos[lane] = p;

    const float* gq = qk  + ((size_t)bi * S + p) * D;
    const float* gv = vin + ((size_t)bi * S + p) * D;

#if HAVE_ASYNC_LDS
    {
        float* lq = &sQK[lane][0];
        float* lv = &sV[lane][0];
#pragma unroll
        for (int i = 0; i < 16; ++i) {
            __builtin_amdgcn_global_load_async_to_lds_b128(AS1V(gq + i * 4), AS3V(lq + i * 4), 0, 0);
            __builtin_amdgcn_global_load_async_to_lds_b128(AS1V(gv + i * 4), AS3V(lv + i * 4), 0, 0);
        }
    }
    ASYNC_WAIT();
    __syncthreads();
    {
        float ss = 0.f;
        for (int f = 0; f < D; ++f) { float x = sQK[lane][f]; ss += x * x; }
        invn[lane] = 1.0f / fmaxf(sqrtf(ss), 1e-12f);
    }
    __syncthreads();
#else
    {
        float ss = 0.f;
        for (int f = 0; f < D; ++f) {
            float x = gq[f]; sQK[lane][f] = x; ss += x * x;
            sV[lane][f] = gv[f];
        }
        invn[lane] = 1.0f / fmaxf(sqrtf(ss), 1e-12f);
    }
    __syncthreads();
#endif

    // A = bq (unnormalized current rows)
    v16h a0, a1;
    for (int e = 0; e < 16; ++e) {
        a0[e] = (_Float16)sQK[nidx][kmapA(e, kh)];
        a1[e] = (_Float16)sQK[nidx][32 + kmapA(e, kh)];
    }

    // dots: two 16x16 output tiles, each K=64 (2 wmma)
    v8f dt[2];
    for (int t = 0; t < 2; ++t) {
        int j = t * 16 + nidx;          // key token index (B lane = column)
        float sc = invn[j];
        v16h b0, b1;
        for (int e = 0; e < 16; ++e) {
            b0[e] = (_Float16)(sQK[j][kmapB(e, kh)]      * sc);
            b1[e] = (_Float16)(sQK[j][32 + kmapB(e, kh)] * sc);
        }
        v8f cacc = {};
        cacc = __builtin_amdgcn_wmma_f32_16x16x32_f16(false, a0, false, b0, (short)0, cacc, false, false);
        cacc = __builtin_amdgcn_wmma_f32_16x16x32_f16(false, a1, false, b1, (short)0, cacc, false, false);
        dt[t] = cacc;
    }

    // scale + self mask (position equality)
    for (int t = 0; t < 2; ++t) {
        int j = t * 16 + nidx;
        for (int r = 0; r < 8; ++r) {
            int m = r + (kh << 3);
            float dv = dt[t][r] * 0.125f;     // d^-0.5 = 1/8
            if (pos[m] == pos[j]) dv = SELF_VAL;
            dt[t][r] = dv;
        }
    }

    // rowwise logsumexp + probs (reduce across the 16 lanes of each half)
    float lse[8];
    for (int r = 0; r < 8; ++r) {
        float mx = fmaxf(dt[0][r], dt[1][r]);
        for (int off = 1; off < 16; off <<= 1) mx = fmaxf(mx, __shfl_xor(mx, off, 32));
        float sm = __expf(dt[0][r] - mx) + __expf(dt[1][r] - mx);
        for (int off = 1; off < 16; off <<= 1) sm += __shfl_xor(sm, off, 32);
        lse[r] = __logf(sm) + mx;
        int m = r + (kh << 3);
        sP[m][nidx]      = (_Float16)__expf(dt[0][r] - lse[r]);
        sP[m][16 + nidx] = (_Float16)__expf(dt[1][r] - lse[r]);
    }
    __syncthreads();

    // bo = probs(16x32) @ bv(32x64): 4 output tiles, K=32 each (1 wmma)
    v16h aP;
    for (int e = 0; e < 16; ++e) aP[e] = sP[nidx][kmapA(e, kh)];
    for (int u = 0; u < 4; ++u) {
        v16h bV;
        for (int e = 0; e < 16; ++e)
            bV[e] = (_Float16)sV[kmapB(e, kh)][u * 16 + nidx];
        v8f cacc = {};
        cacc = __builtin_amdgcn_wmma_f32_16x16x32_f16(false, aP, false, bV, (short)0, cacc, false, false);
        for (int r = 0; r < 8; ++r) {
            int m = r + (kh << 3);
            so[((size_t)bi * HS + c * 16 + m) * D + u * 16 + nidx] = cacc[r];
        }
    }
    if (nidx == 0) {
        for (int r = 0; r < 8; ++r) {
            int m = r + (kh << 3);
            slog[(size_t)bi * HS + c * 16 + m] = lse[r];
        }
    }
}

// ---------------------------------------------------------------------------
// 4) Un-sort + combine across hashes: out = sum_h softmax(logits_h) * o_h
// ---------------------------------------------------------------------------
__global__ void __launch_bounds__(64)
combine_kernel(const int* __restrict__ inv, const float* __restrict__ so,
               const float* __restrict__ slog, float* __restrict__ out)
{
    __shared__ float lg[NH];
    __shared__ int   sl[NH];
    int bp = blockIdx.x;
    int p  = bp & (S - 1);
    int bi = bp >> 13;
    int dd = threadIdx.x;
    if (dd < NH) {
        int j = inv[(size_t)bi * HS + dd * S + p];
        sl[dd] = j;
        lg[dd] = slog[(size_t)bi * HS + j];
    }
    __syncthreads();
    // prefetch the 8 gathered rows (global_prefetch_b8)
    for (int g = 0; g < NH; ++g)
        __builtin_prefetch(&so[((size_t)bi * HS + sl[g]) * D + dd], 0, 0);
    float mx = -3.4e38f;
    for (int g = 0; g < NH; ++g) mx = fmaxf(mx, lg[g]);
    float sm = 0.f;
    for (int g = 0; g < NH; ++g) sm += __expf(lg[g] - mx);
    float l = __logf(sm) + mx;
    float acc = 0.f;
    for (int g = 0; g < NH; ++g) {
        float w = __expf(lg[g] - l);
        acc += w * so[((size_t)bi * HS + sl[g]) * D + dd];
    }
    out[((size_t)bi * S + p) * D + dd] = acc;
}

// ---------------------------------------------------------------------------
extern "C" void kernel_launch(void* const* d_in, const int* in_sizes, int n_in,
                              void* d_out, int out_size, void* d_ws, size_t ws_size,
                              hipStream_t stream)
{
    (void)in_sizes; (void)n_in; (void)out_size; (void)ws_size;
    const float* qk  = (const float*)d_in[0];
    const float* v   = (const float*)d_in[1];
    const float* rot = (const float*)d_in[2];
    float* out = (float*)d_out;
    int* out_buckets = (int*)(out + (size_t)Bb * S * D);   // second tuple output

    char* ws = (char*)d_ws;
    size_t off = 0;
    auto alloc = [&](size_t bytes) -> char* {
        char* pp = ws + off; off += (bytes + 255) & ~(size_t)255; return pp;
    };
    int*   keys    = (int*)  alloc((size_t)Bb * HS * 4);
    int*   hist    = (int*)  alloc((size_t)Bb * BINS * 4);
    int*   sticker = (int*)  alloc((size_t)Bb * HS * 4);
    int*   inv     = (int*)  alloc((size_t)Bb * HS * 4);
    float* slog    = (float*)alloc((size_t)Bb * HS * 4);
    float* so      = (float*)alloc((size_t)Bb * HS * D * 4);   // 128 MB

    hash_kernel<<<Bb * NH * (S / 16), 32, 0, stream>>>(qk, rot, keys, out_buckets);

    int nh = Bb * BINS;
    zero_hist<<<(nh + 255) / 256, 256, 0, stream>>>(hist);
    int nt = Bb * HS;
    hist_kernel<<<(nt + 255) / 256, 256, 0, stream>>>(keys, hist);
    scan_kernel<<<Bb, 256, 0, stream>>>(hist);
    scatter_kernel<<<(nt + 255) / 256, 256, 0, stream>>>(keys, hist, sticker, inv);

    attn_kernel<<<Bb * NC, 32, 0, stream>>>(qk, v, sticker, so, slog);
    combine_kernel<<<Bb * S, 64, 0, stream>>>(inv, so, slog, out);
}
